// TopologyAwareGCNEncoder_67869073211462
// MI455X (gfx1250) — matrix operand order repaired and checked by reading, not compiled, not run on hardware
//
#include <hip/hip_runtime.h>

// ---------------------------------------------------------------------------
// TopologyAwareGCNEncoder for MI455X (gfx1250, wave32, WMMA f16 16x16x32).
//
//  out = relu(relu(A(A X W1 + b1 relu) W2 + b2) . Wm1 + bm1) . Wm2 + bm2
//  using (A h) W == A (h W): aggregations are cheap sparse VALU passes,
//  all heavy math is v_wmma_f32_16x16x32_f16.
//
//  Kernel 1 (prep): pack W2 / Wm1 / Wm2 (f32 -> f16) into the exact WMMA
//     B-operand lane layout in workspace (~584 KB, L2 resident).
//  Kernel 2 (fused): one 4-wave block per 64 samples, everything staged in
//     LDS (276 KB dynamic), 712 WMMA per wave per 16 samples.
// ---------------------------------------------------------------------------

typedef __attribute__((ext_vector_type(16))) _Float16      v16h;
typedef __attribute__((ext_vector_type(8)))  float         v8f;
typedef __attribute__((ext_vector_type(2)))  _Float16      h2v;
typedef __attribute__((ext_vector_type(4)))  unsigned int  u32x4;
typedef __attribute__((ext_vector_type(4)))  float         f32x4;

#define ROWS_PER_WAVE   272      // 16 samples * 17 nodes
#define SAMPLES_PER_BLK 64
#define KC_MLP1         34       // 1088 / 32

// COCO-17 skeleton adjacency (incl. self loops), CSR over 17 nodes.
__device__ __constant__ int NBR_PTR[18] =
  {0,5,8,11,13,15,19,23,26,29,31,33,36,39,42,45,47,49};
__device__ __constant__ int NBR_IDX[49] = {
  0,1,2,5,6,   1,0,3,   2,0,4,   3,1,   4,2,
  5,0,7,11,    6,0,8,12, 7,5,9,  8,6,10, 9,7, 10,8,
  11,5,13, 12,6,14, 13,11,15, 14,12,16, 15,13, 16,14 };

// LDS const-region offsets (in floats): adj 289 | W1 128 | b1 64 | b2 64 | bm1 256 | bm2 64
#define C_ADJ  0
#define C_W1   296
#define C_B1   424
#define C_B2   488
#define C_BM1  552
#define C_BM2  808
#define CONST_BYTES 3584                 // 896 floats, 16B aligned
#define WAVE_LDS_BYTES 69632             // bufA(34816) + bufB(34816)
#define TOTAL_LDS (CONST_BYTES + 4 * WAVE_LDS_BYTES)   // 282112 B < 320 KB

// ---- WMMA helpers ---------------------------------------------------------

__device__ __forceinline__ v8f wmma16(v16h a, v16h b, v8f c) {
  return __builtin_amdgcn_wmma_f32_16x16x32_f16(false, a, false, b,
                                                (short)0, c, false, false);
}

// A-operand (16x32 f16): lane = half*16+l16 reads row l16,
//   VGPR0-7 = K half*16..half*16+15 contiguous
__device__ __forceinline__ v16h load_a_lds(const _Float16* rowk, int half) {
  union { u32x4 q[2]; v16h v; } u;
  const _Float16* p = rowk + half * 16;
  u.q[0] = *(const u32x4*)(p);
  u.q[1] = *(const u32x4*)(p + 8);
  return u.v;
}

// B-operand pre-packed: each lane owns 16 contiguous f16 (32 B).
__device__ __forceinline__ v16h load_b_glb(const _Float16* p) {
  union { u32x4 q[2]; v16h v; } u;
  u.q[0] = *(const u32x4*)(p);
  u.q[1] = *(const u32x4*)(p + 8);
  return u.v;
}

// ---- Kernel 1: pack weights into WMMA B-operand layout --------------------
// B 32x16 f16 layout: lane L holds N = nt*16 + L%16,
//   K = kc*32 + (L/16)*16 + t, t = 0..15 contiguous per lane.
__global__ void pack_weights_kernel(const float* __restrict__ W2,
                                    const float* __restrict__ Wm1,
                                    const float* __restrict__ Wm2,
                                    _Float16* __restrict__ pW2,
                                    _Float16* __restrict__ pWm1,
                                    _Float16* __restrict__ pWm2) {
  int tid = blockIdx.x * blockDim.x + threadIdx.x;
  int stride = gridDim.x * blockDim.x;

  // W2 (64,64): KC=2, NT=4
  for (int idx = tid; idx < 2 * 4 * 32 * 16; idx += stride) {
    int t = idx & 15, lane = (idx >> 4) & 31, nt = (idx >> 9) & 3, kc = idx >> 11;
    int k = kc * 32 + (lane >> 4) * 16 + t;
    int n = nt * 16 + (lane & 15);
    pW2[idx] = (_Float16)W2[k * 64 + n];
  }
  // Wm1 (1088,256): KC=34, NT=16
  for (int idx = tid; idx < KC_MLP1 * 16 * 32 * 16; idx += stride) {
    int t = idx & 15, lane = (idx >> 4) & 31, nt = (idx >> 9) & 15, kc = idx >> 13;
    int k = kc * 32 + (lane >> 4) * 16 + t;
    int n = nt * 16 + (lane & 15);
    pWm1[idx] = (_Float16)Wm1[k * 256 + n];
  }
  // Wm2 (256,50) padded to N=64: KC=8, NT=4
  for (int idx = tid; idx < 8 * 4 * 32 * 16; idx += stride) {
    int t = idx & 15, lane = (idx >> 4) & 31, nt = (idx >> 9) & 3, kc = idx >> 11;
    int k = kc * 32 + (lane >> 4) * 16 + t;
    int n = nt * 16 + (lane & 15);
    pWm2[idx] = (n < 50) ? (_Float16)Wm2[k * 50 + n] : (_Float16)0.f;
  }
}

// ---- Kernel 2: fused GCN + MLP -------------------------------------------

extern __shared__ f32x4 smem_raw[];

__global__ void __launch_bounds__(128, 1)
gcn_mlp_fused_kernel(const float* __restrict__ x,
                     const float* __restrict__ adj,
                     const float* __restrict__ W1,
                     const float* __restrict__ b1,
                     const float* __restrict__ b2,
                     const float* __restrict__ bm1,
                     const float* __restrict__ bm2,
                     const _Float16* __restrict__ pW2,
                     const _Float16* __restrict__ pWm1,
                     const _Float16* __restrict__ pWm2,
                     float* __restrict__ out) {
  char*  smem = (char*)smem_raw;
  float* cst  = (float*)smem;
  const int tid  = threadIdx.x;
  const int w    = tid >> 5;
  const int lane = tid & 31;
  const int l16  = lane & 15;
  const int half = lane >> 4;

  _Float16* bufA = (_Float16*)(smem + CONST_BYTES + w * WAVE_LDS_BYTES); // h1 -> h2
  _Float16* bufB = bufA + 17408;                                         // g  -> m1

  // ---- stage constants into LDS ----
  for (int i = tid; i < 289; i += 128) cst[C_ADJ + i] = adj[i];
  for (int i = tid; i < 128; i += 128) cst[C_W1 + i]  = W1[i];
  for (int i = tid; i < 64;  i += 128) { cst[C_B1 + i] = b1[i]; cst[C_B2 + i] = b2[i]; }
  for (int i = tid; i < 256; i += 128) cst[C_BM1 + i] = bm1[i];
  for (int i = tid; i < 64;  i += 128) cst[C_BM2 + i] = (i < 50) ? bm2[i] : 0.f;
  __syncthreads();

  const int sbase = blockIdx.x * SAMPLES_PER_BLK + w * 16;  // wave's 16 samples

  // ---- Phase 1: h1 = relu((A X) W1 + b1) -> bufA (272 x 64 f16) ----
  for (int p = lane; p < ROWS_PER_WAVE; p += 32) {
    int s = p / 17, n = p - s * 17;
    const float* xs = x + (size_t)(sbase + s) * 34;
    float ax0 = 0.f, ax1 = 0.f;
    int e0 = NBR_PTR[n], e1 = NBR_PTR[n + 1];
    for (int e = e0; e < e1; ++e) {
      int j = NBR_IDX[e];
      float a = cst[C_ADJ + n * 17 + j];
      ax0 = fmaf(a, xs[2 * j],     ax0);
      ax1 = fmaf(a, xs[2 * j + 1], ax1);
    }
    _Float16* dst = bufA + p * 64;
    #pragma unroll 8
    for (int f = 0; f < 64; f += 2) {
      float v0 = fmaxf(fmaf(ax0, cst[C_W1 + f],     fmaf(ax1, cst[C_W1 + 64 + f],     cst[C_B1 + f])),     0.f);
      float v1 = fmaxf(fmaf(ax0, cst[C_W1 + f + 1], fmaf(ax1, cst[C_W1 + 64 + f + 1], cst[C_B1 + f + 1])), 0.f);
      h2v hv; hv[0] = (_Float16)v0; hv[1] = (_Float16)v1;
      *(h2v*)(dst + f) = hv;
    }
  }
  __syncthreads();

  // ---- Phase 2: g = A h1 -> bufB (272 x 64 f16), sparse aggregation ----
  for (int p = lane; p < ROWS_PER_WAVE; p += 32) {
    int s = p / 17, n = p - s * 17;
    const _Float16* src = bufA + s * 1088;
    _Float16* dst = bufB + p * 64;
    int e0 = NBR_PTR[n], e1 = NBR_PTR[n + 1];
    for (int f2 = 0; f2 < 32; ++f2) {
      float a0 = 0.f, a1 = 0.f;
      for (int e = e0; e < e1; ++e) {
        int j = NBR_IDX[e];
        float a = cst[C_ADJ + n * 17 + j];
        h2v hv = *(const h2v*)(src + j * 64 + 2 * f2);
        a0 = fmaf(a, (float)hv[0], a0);
        a1 = fmaf(a, (float)hv[1], a1);
      }
      h2v o; o[0] = (_Float16)a0; o[1] = (_Float16)a1;
      *(h2v*)(dst + 2 * f2) = o;
    }
  }
  __syncthreads();

  // ---- Phase 3: h2 = relu(g W2 + b2) -> bufA (WMMA, 17 M-tiles) ----
  {
    v16h Bw[8];
    #pragma unroll
    for (int i = 0; i < 8; ++i)
      Bw[i] = load_b_glb(pW2 + (i * 32 + lane) * 16);

    for (int mt = 0; mt < 17; ++mt) {
      const _Float16* arow = bufB + (mt * 16 + l16) * 64;
      v16h a0 = load_a_lds(arow,      half);
      v16h a1 = load_a_lds(arow + 32, half);
      v8f acc[4];
      #pragma unroll
      for (int nt = 0; nt < 4; ++nt) {
        v8f z = {0.f,0.f,0.f,0.f,0.f,0.f,0.f,0.f};
        z = wmma16(a0, Bw[nt],     z);
        z = wmma16(a1, Bw[4 + nt], z);
        acc[nt] = z;
      }
      #pragma unroll
      for (int nt = 0; nt < 4; ++nt) {
        int f = nt * 16 + l16;
        float bias = cst[C_B2 + f];
        #pragma unroll
        for (int vi = 0; vi < 8; ++vi) {
          int rr = mt * 16 + vi + half * 8;
          bufA[rr * 64 + f] = (_Float16)fmaxf(acc[nt][vi] + bias, 0.f);
        }
      }
    }
  }
  __syncthreads();

  // ---- Phase 4: m1 = relu(h2 Wm1 + bm1) -> bufB (16 x 256 f16) ----
  {
    const _Float16* arowbase = bufA + l16 * 1088;   // sample row, 1088 f16
    for (int ntc = 0; ntc < 4; ++ntc) {
      v8f acc[4];
      #pragma unroll
      for (int n4 = 0; n4 < 4; ++n4) {
        v8f z = {0.f,0.f,0.f,0.f,0.f,0.f,0.f,0.f};
        acc[n4] = z;
      }
      for (int kc = 0; kc < KC_MLP1; ++kc) {
        v16h a = load_a_lds(arowbase + kc * 32, half);
        #pragma unroll
        for (int n4 = 0; n4 < 4; ++n4) {
          int nt = ntc * 4 + n4;
          v16h b = load_b_glb(pWm1 + ((kc * 16 + nt) * 32 + lane) * 16);
          acc[n4] = wmma16(a, b, acc[n4]);
        }
      }
      #pragma unroll
      for (int n4 = 0; n4 < 4; ++n4) {
        int f = (ntc * 4 + n4) * 16 + l16;
        float bias = cst[C_BM1 + f];
        #pragma unroll
        for (int vi = 0; vi < 8; ++vi) {
          int rr = vi + half * 8;
          bufB[rr * 256 + f] = (_Float16)fmaxf(acc[n4][vi] + bias, 0.f);
        }
      }
    }
  }
  __syncthreads();

  // ---- Phase 5: out = m1 Wm2 + bm2 (N padded 50->64), f32 store ----
  {
    v8f acc[4];
    #pragma unroll
    for (int nt = 0; nt < 4; ++nt) {
      v8f z = {0.f,0.f,0.f,0.f,0.f,0.f,0.f,0.f};
      acc[nt] = z;
    }
    for (int kc = 0; kc < 8; ++kc) {
      v16h a = load_a_lds(bufB + l16 * 256 + kc * 32, half);
      #pragma unroll
      for (int nt = 0; nt < 4; ++nt) {
        v16h b = load_b_glb(pWm2 + ((kc * 4 + nt) * 32 + lane) * 16);
        acc[nt] = wmma16(a, b, acc[nt]);
      }
    }
    #pragma unroll
    for (int nt = 0; nt < 4; ++nt) {
      int n = nt * 16 + l16;
      if (n < 50) {
        float bias = cst[C_BM2 + n];
        #pragma unroll
        for (int vi = 0; vi < 8; ++vi) {
          int gs = sbase + vi + half * 8;
          out[(size_t)gs * 50 + n] = acc[nt][vi] + bias;
        }
      }
    }
  }
}

// ---- host launch ----------------------------------------------------------

extern "C" void kernel_launch(void* const* d_in, const int* in_sizes, int n_in,
                              void* d_out, int out_size, void* d_ws, size_t ws_size,
                              hipStream_t stream) {
  const float* x   = (const float*)d_in[0];
  const float* adj = (const float*)d_in[1];
  const float* W1  = (const float*)d_in[2];
  const float* b1  = (const float*)d_in[3];
  const float* W2  = (const float*)d_in[4];
  const float* b2  = (const float*)d_in[5];
  const float* Wm1 = (const float*)d_in[6];
  const float* bm1 = (const float*)d_in[7];
  const float* Wm2 = (const float*)d_in[8];
  const float* bm2 = (const float*)d_in[9];
  float* out = (float*)d_out;

  const int B = in_sizes[0] / 34;   // (B,17,2)

  // Workspace: packed f16 B-operands (needs ~584 KB).
  _Float16* pW2  = (_Float16*)d_ws;                 //   4096 f16
  _Float16* pWm1 = pW2 + 4096;                      // 278528 f16
  _Float16* pWm2 = pWm1 + 278528;                   //  16384 f16

  pack_weights_kernel<<<128, 256, 0, stream>>>(W2, Wm1, Wm2, pW2, pWm1, pWm2);

  const int blocks = B / SAMPLES_PER_BLK;           // 1024 for B=65536
  gcn_mlp_fused_kernel<<<blocks, 128, TOTAL_LDS, stream>>>(
      x, adj, W1, b1, b2, bm1, bm2, pW2, pWm1, pWm2, out);
}
